// GCN_88948772700498
// MI455X (gfx1250) — matrix operand order repaired and checked by reading, not compiled, not run on hardware
//
#include <hip/hip_runtime.h>
#include <hip/hip_bf16.h>

typedef __attribute__((ext_vector_type(8)))  float   v8f;
typedef __attribute__((ext_vector_type(16))) __bf16  v16bf;
typedef __attribute__((ext_vector_type(8)))  __bf16  v8bf;

#define N_NODES 8192
#define F       128
#define MT      32     // output rows per workgroup
#define KC      128    // K chunk staged in LDS
#define ASTR    136    // padded LDS row stride (ushorts) -> bank-conflict-free A frags

__device__ __forceinline__ unsigned short f32_to_bf16_rne(float f) {
    unsigned int u = __float_as_uint(f);
    u += 0x7FFFu + ((u >> 16) & 1u);
    return (unsigned short)(u >> 16);
}

// hsT block-interleaved index: element (k, n) of B matrix
__device__ __forceinline__ size_t hsT_idx(int k, int n) {
    return ((size_t)(k >> 5) * F + n) * 32 + (k & 31);
}

// ---------------- Kernel A: rowsum of (adj + I) -> r_inv_sqrt ----------------
__global__ void gcn_rowsum_kernel(const float* __restrict__ adj,
                                  float* __restrict__ rinv) {
    __shared__ float red[256];
    const int row = blockIdx.x;
    const int t = threadIdx.x;
    const float* arow = adj + (size_t)row * N_NODES;
    float s = 0.f;
    for (int j = t; j < N_NODES; j += 256) s += arow[j];
    red[t] = s;
    __syncthreads();
    for (int off = 128; off > 0; off >>= 1) {
        if (t < off) red[t] += red[t + off];
        __syncthreads();
    }
    if (t == 0) {
        float rs = red[0] + 1.0f;   // +1 from the self-loop
        rinv[row] = (rs > 0.f) ? (1.0f / sqrtf(rs)) : 0.f;
    }
}

// ---------- Kernel B: hs = rinv * (x @ W^T + b); store f32 + bf16(T) ----------
__global__ void gcn_feature_kernel(const float* __restrict__ x,
                                   const float* __restrict__ W,
                                   const float* __restrict__ bias,
                                   const float* __restrict__ rinv,
                                   float* __restrict__ hs,
                                   unsigned short* __restrict__ hsT) {
    __shared__ float xs[16][F];
    const int t = threadIdx.x;              // 256 threads
    const int row0 = blockIdx.x * 16;
    // stage 16 rows of x (16*128 floats, 8 per thread)
    for (int i = 0; i < 8; ++i) {
        int idx = t + 256 * i;
        int r = idx >> 7, c = idx & 127;
        xs[r][c] = x[(size_t)(row0 + r) * F + c];
    }
    __syncthreads();
    const int o = t & 127;                  // output feature
    const int rbase = (t >> 7) * 8;         // rows rbase..rbase+7
    float acc[8];
    const float bv = bias[o];
    for (int j = 0; j < 8; ++j) acc[j] = bv;
    for (int k = 0; k < F; ++k) {
        float w = W[(size_t)o * F + k];     // weight [out, in]
        for (int j = 0; j < 8; ++j) acc[j] = fmaf(xs[rbase + j][k], w, acc[j]);
    }
    for (int j = 0; j < 8; ++j) {
        int i = row0 + rbase + j;
        float v = rinv[i] * acc[j];
        hs[(size_t)i * F + o] = v;
        hsT[hsT_idx(i, o)] = f32_to_bf16_rne(v);
    }
}

// -------- Kernel C: out = rinv ⊙ (adj @ hs + hs)  via bf16 WMMA --------
__global__ void gcn_aggregate_kernel(const float* __restrict__ adj,
                                     const float* __restrict__ hs,
                                     const unsigned short* __restrict__ hsT,
                                     const float* __restrict__ rinv,
                                     float* __restrict__ out) {
    __shared__ unsigned short As[MT * ASTR];   // bf16 bits of adj tile
    const int t = threadIdx.x;                 // 256 = 8 waves
    const int lane = t & 31;
    const int wave = t >> 5;                   // 0..7 -> N slice
    const int m_base = blockIdx.x * MT;
    const int n0 = wave * 16;
    const int mrow = lane & 15;                // M (A) / N (B,C) within tile
    const int hlf = lane >> 4;

    v8f c0 = {};  // rows m_base + 0..15
    v8f c1 = {};  // rows m_base + 16..31

    for (int k0 = 0; k0 < N_NODES; k0 += KC) {
        __syncthreads();
        // cooperative stage: MT x KC floats of adj -> bf16 in LDS
        for (int i4 = 0; i4 < 4; ++i4) {
            int idx = t * 16 + i4 * 4;       // 4096 elements total
            int r = idx >> 7;                // KC == 128
            int c = idx & 127;
            const float4 v = *(const float4*)(adj + (size_t)(m_base + r) * N_NODES + k0 + c);
            unsigned int p0 = (unsigned int)f32_to_bf16_rne(v.x) |
                              ((unsigned int)f32_to_bf16_rne(v.y) << 16);
            unsigned int p1 = (unsigned int)f32_to_bf16_rne(v.z) |
                              ((unsigned int)f32_to_bf16_rne(v.w) << 16);
            *(unsigned int*)(&As[r * ASTR + c])     = p0;
            *(unsigned int*)(&As[r * ASTR + c + 2]) = p1;
        }
        __syncthreads();

        for (int kk = 0; kk < KC; kk += 32) {
            // B fragment: lane holds column (n0+mrow), K = k0+kk+16*hlf .. +15 (contiguous)
            const v16bf bfrag =
                *(const v16bf*)(hsT + hsT_idx(k0 + kk, n0 + mrow) + 16 * hlf);

            // A fragment rows m_base+mrow: lo = K kk+16*hlf..+7, hi = K kk+8+16*hlf..+7
            const v8bf alo0 = *(const v8bf*)(&As[mrow * ASTR + kk + 16 * hlf]);
            const v8bf ahi0 = *(const v8bf*)(&As[mrow * ASTR + kk + 8 + 16 * hlf]);
            v16bf a0 = __builtin_shufflevector(alo0, ahi0,
                        0,1,2,3,4,5,6,7,8,9,10,11,12,13,14,15);
            c0 = __builtin_amdgcn_wmma_f32_16x16x32_bf16(
                     false, a0, false, bfrag, (short)0, c0, false, false);

            const v8bf alo1 = *(const v8bf*)(&As[(16 + mrow) * ASTR + kk + 16 * hlf]);
            const v8bf ahi1 = *(const v8bf*)(&As[(16 + mrow) * ASTR + kk + 8 + 16 * hlf]);
            v16bf a1 = __builtin_shufflevector(alo1, ahi1,
                        0,1,2,3,4,5,6,7,8,9,10,11,12,13,14,15);
            c1 = __builtin_amdgcn_wmma_f32_16x16x32_bf16(
                     false, a1, false, bfrag, (short)0, c1, false, false);
        }
    }

    // epilogue: D layout is VGPR r -> M = r + 8*hlf, N = lane&15
    const int nn = n0 + mrow;
    for (int r = 0; r < 8; ++r) {
        int m = m_base + r + 8 * hlf;
        out[(size_t)m * F + nn] = rinv[m] * (c0[r] + hs[(size_t)m * F + nn]);
        int m2 = m + 16;
        out[(size_t)m2 * F + nn] = rinv[m2] * (c1[r] + hs[(size_t)m2 * F + nn]);
    }
}

extern "C" void kernel_launch(void* const* d_in, const int* in_sizes, int n_in,
                              void* d_out, int out_size, void* d_ws, size_t ws_size,
                              hipStream_t stream) {
    const float* x   = (const float*)d_in[0];
    const float* adj = (const float*)d_in[1];
    const float* W   = (const float*)d_in[2];
    const float* b   = (const float*)d_in[3];
    float* out = (float*)d_out;

    // workspace layout: rinv[8192] f32 | hs[8192*128] f32 | hsT[128*8192] bf16
    float* rinv = (float*)d_ws;
    float* hs   = rinv + N_NODES;
    unsigned short* hsT = (unsigned short*)(hs + (size_t)N_NODES * F);

    gcn_rowsum_kernel<<<N_NODES, 256, 0, stream>>>(adj, rinv);
    gcn_feature_kernel<<<N_NODES / 16, 256, 0, stream>>>(x, W, b, rinv, hs, hsT);
    gcn_aggregate_kernel<<<N_NODES / MT, 256, 0, stream>>>(adj, hs, hsT, rinv, out);
}
